// GroupedQueryAttention_79216376807438
// MI455X (gfx1250) — matrix operand order repaired and checked by reading, not compiled, not run on hardware
//
#include <hip/hip_runtime.h>

typedef unsigned short u16;
typedef __attribute__((ext_vector_type(16))) __bf16 v16bf;
typedef __attribute__((ext_vector_type(8)))  float  v8f;

#define DIMS 2048
#define BSZ  2
#define SEQ  1024
#define BS   (BSZ * SEQ)   // 2048 rows
#define QH   32
#define KH   8
#define HD   64
#define KVD  (KH * HD)     // 512
#define REPF (QH / KH)     // 4

// ---------------------------------------------------------------- utilities
__device__ __forceinline__ u16 f2bf(float f) {
  union { float f; unsigned u; } v; v.f = f;
  unsigned r = v.u + 0x7FFFu + ((v.u >> 16) & 1u);  // round-nearest-even
  return (u16)(r >> 16);
}

struct Frag { uint4 lo, hi; };
__device__ __forceinline__ v16bf load_frag(const u16* p0, const u16* p1) {
  Frag f;
  f.lo = *(const uint4*)p0;
  f.hi = *(const uint4*)p1;
  return __builtin_bit_cast(v16bf, f);
}

// Generic LDS pointers carry the aperture in the high 32 bits and the DS
// offset in addr[31:0] (ISA 10.2) -> truncation yields the LDS byte address.
__device__ __forceinline__ unsigned lds_addr(const void* p) {
  return (unsigned)(uintptr_t)p;
}

// CDNA5 async memory->LDS copy (ASYNCcnt-tracked, no VGPR round trip).
__device__ __forceinline__ void async_ld_b128(unsigned lds, const void* gptr) {
  asm volatile("global_load_async_to_lds_b128 %0, %1, off"
               :: "v"(lds), "v"(gptr) : "memory");
}
__device__ __forceinline__ void wait_async0() {
  asm volatile("s_wait_asynccnt 0" ::: "memory");
}

// ---------------------------------------------------------------- fp32 -> bf16
__global__ void cvt_f32_bf16(const float* __restrict__ in, u16* __restrict__ out, int n) {
  for (int i = blockIdx.x * blockDim.x + threadIdx.x; i < n; i += gridDim.x * blockDim.x)
    out[i] = f2bf(in[i]);
}

// ---------------------------------------------------------------- bf16 WMMA GEMM
// C[M,N] = A[M,K] * B[K,N] + bias ; block tile 128x64, K-step 32, 256 threads.
// Double-buffered async staging: tile k+1 streams to LDS while WMMAs chew tile k.
#define BM 128
#define BN 64
#define BKK 32

__global__ __launch_bounds__(256)
void gemm_bf16(const u16* __restrict__ A, const u16* __restrict__ Bm,
               const float* __restrict__ bias,
               float* __restrict__ Cf, u16* __restrict__ Cbf,
               int K, int lda, int ldb, int ldc) {
  __shared__ u16 As[2][BM][BKK + 8];   // row stride 80B (16B aligned)
  __shared__ u16 Bs[2][BKK][BN + 8];   // row stride 144B (16B aligned)

  const int tid  = threadIdx.x;
  const int lane = tid & 31;
  const int wid  = tid >> 5;
  const int hf   = lane >> 4;          // lane half: K-range (A) / M+8 (C)
  const int l16  = lane & 15;

  const int blockM = blockIdx.y * BM;
  const int blockN = blockIdx.x * BN;
  const int wm = wid & 3;              // 4 waves along M (32 rows each)
  const int wn = wid >> 2;             // 2 waves along N (32 cols each)

  // per-thread staging chunks (A: 2x16B, B: 1x16B)
  const int ar0 = tid >> 2,         ac0 = (tid & 3) * 8;
  const int ar1 = (tid + 256) >> 2, ac1 = (tid & 3) * 8;  // (tid+256)&3 == tid&3
  const int br  = tid >> 3,         bc  = (tid & 7) * 8;

  auto stage = [&](int buf, int k0) {
    async_ld_b128(lds_addr(&As[buf][ar0][ac0]), A + (size_t)(blockM + ar0) * lda + k0 + ac0);
    async_ld_b128(lds_addr(&As[buf][ar1][ac1]), A + (size_t)(blockM + ar1) * lda + k0 + ac1);
    async_ld_b128(lds_addr(&Bs[buf][br][bc]),   Bm + (size_t)(k0 + br) * ldb + blockN + bc);
  };

  v8f acc[2][2];
  acc[0][0] = {}; acc[0][1] = {}; acc[1][0] = {}; acc[1][1] = {};

  stage(0, 0);
  wait_async0();
  __syncthreads();

  int cur = 0;
  for (int k0 = 0; k0 < K; k0 += BKK) {
    if (k0 + BKK < K) stage(cur ^ 1, k0 + BKK);   // overlap with WMMA below

    // fragments (ISA 7.12.2 layouts)
    v16bf a0, a1, b0, b1;
    {
      const u16* ap0 = &As[cur][wm * 32 + l16][0];
      const u16* ap1 = &As[cur][wm * 32 + 16 + l16][0];
      a0 = load_frag(ap0 + hf * 8, ap0 + 16 + hf * 8);
      a1 = load_frag(ap1 + hf * 8, ap1 + 16 + hf * 8);
      u16 tb0[16], tb1[16];
      #pragma unroll
      for (int s = 0; s < 16; s++) {
        const int kk = (s < 8) ? hf * 8 + s : 16 + hf * 8 + (s - 8);
        tb0[s] = Bs[cur][kk][wn * 32 + l16];
        tb1[s] = Bs[cur][kk][wn * 32 + 16 + l16];
      }
      b0 = __builtin_bit_cast(v16bf, tb0);
      b1 = __builtin_bit_cast(v16bf, tb1);
    }
    acc[0][0] = __builtin_amdgcn_wmma_f32_16x16x32_bf16(false, a0, false, b0, (short)0, acc[0][0], false, false);
    acc[0][1] = __builtin_amdgcn_wmma_f32_16x16x32_bf16(false, a0, false, b1, (short)0, acc[0][1], false, false);
    acc[1][0] = __builtin_amdgcn_wmma_f32_16x16x32_bf16(false, a1, false, b0, (short)0, acc[1][0], false, false);
    acc[1][1] = __builtin_amdgcn_wmma_f32_16x16x32_bf16(false, a1, false, b1, (short)0, acc[1][1], false, false);

    wait_async0();        // next tile resident in LDS
    __syncthreads();      // all waves done with 'cur' and done staging
    cur ^= 1;
  }

  // epilogue: C layout M = vgpr + 8*half, N = lane%16
  #pragma unroll
  for (int i = 0; i < 2; i++) {
    #pragma unroll
    for (int j = 0; j < 2; j++) {
      const int col = blockN + wn * 32 + j * 16 + l16;
      const float bv = bias ? bias[col] : 0.0f;
      #pragma unroll
      for (int r = 0; r < 8; r++) {
        const int row = blockM + wm * 32 + i * 16 + r + 8 * hf;
        const float v = acc[i][j][r] + bv;
        if (Cbf) Cbf[(size_t)row * ldc + col] = f2bf(v);
        else     Cf [(size_t)row * ldc + col] = v;
      }
    }
  }
}

// ---------------------------------------------------------------- flash GQA
// grid = (SEQ/128, QH, B); 256 threads = 8 waves x 16 query rows.
__global__ __launch_bounds__(256)
void gqa_flash(const u16* __restrict__ Qp, const u16* __restrict__ Kp,
               const u16* __restrict__ Vp, u16* __restrict__ Ao) {
  __shared__ u16 Kt[HD][32 + 8];      // K tile transposed: [d][key]
  __shared__ u16 Vs[32][HD + 8];      // V tile: [key][d]
  __shared__ u16 Ps[8][16][32 + 8];   // per-wave P repack buffer

  const int tid  = threadIdx.x;
  const int lane = tid & 31;
  const int wid  = tid >> 5;
  const int hf   = lane >> 4;
  const int l16  = lane & 15;

  const int qt  = blockIdx.x;
  const int h   = blockIdx.y;
  const int b   = blockIdx.z;
  const int kvh = h / REPF;
  const int qb  = qt * 128;
  const int qrow0 = qb + wid * 16;

  const u16* Qbase = Qp + (size_t)b * SEQ * DIMS + h * HD;
  const u16* Kbase = Kp + (size_t)b * SEQ * KVD + kvh * HD;
  const u16* Vbase = Vp + (size_t)b * SEQ * KVD + kvh * HD;

  // Q fragment (16x64 -> 2 A fragments), loaded once per wave
  v16bf aq[2];
  #pragma unroll
  for (int c = 0; c < 2; c++) {
    const u16* qp = Qbase + (size_t)(qrow0 + l16) * DIMS + c * 32;
    aq[c] = load_frag(qp + hf * 8, qp + 16 + hf * 8);
  }

  v8f o[4]; o[0] = {}; o[1] = {}; o[2] = {}; o[3] = {};
  float m[8], l[8];
  #pragma unroll
  for (int r = 0; r < 8; r++) { m[r] = -1e30f; l[r] = 0.0f; }

  const int key = tid >> 3;            // staging: this thread's key row
  const int d0  = (tid & 7) * 8;       // and 8 contiguous d

  const int nkt = (qb + 128) / 32;     // causal: tiles up to block's last row
  for (int kt = 0; kt < nkt; kt++) {
    const int kb = kt * 32;
    // V tile: straight copy -> async memory->LDS
    async_ld_b128(lds_addr(&Vs[key][d0]), Vbase + (size_t)(kb + key) * KVD + d0);
    // K tile: transposed on the way in (through VGPRs)
    {
      uint4 kq = *(const uint4*)(Kbase + (size_t)(kb + key) * KVD + d0);
      u16 tmp[8]; *(uint4*)tmp = kq;
      #pragma unroll
      for (int e = 0; e < 8; e++) Kt[d0 + e][key] = tmp[e];
    }
    wait_async0();
    __syncthreads();

    // scores: S = Q(16x64) x K^T(64x32) -> two 16x16 C fragments
    v8f cs[2]; cs[0] = {}; cs[1] = {};
    #pragma unroll
    for (int nt = 0; nt < 2; nt++) {
      #pragma unroll
      for (int c = 0; c < 2; c++) {
        u16 tk[16];                                    // B: N=l16(key), K=d in table order
        #pragma unroll
        for (int s = 0; s < 16; s++) {
          const int kk = (s < 8) ? hf * 8 + s : 16 + hf * 8 + (s - 8);
          tk[s] = Kt[c * 32 + kk][nt * 16 + l16];
        }
        v16bf bkf = __builtin_bit_cast(v16bf, tk);
        cs[nt] = __builtin_amdgcn_wmma_f32_16x16x32_bf16(false, aq[c], false, bkf, (short)0, cs[nt], false, false);
      }
    }

    // scale + causal mask (in C layout)
    #pragma unroll
    for (int nt = 0; nt < 2; nt++) {
      #pragma unroll
      for (int r = 0; r < 8; r++) {
        const int qrow = qrow0 + r + 8 * hf;
        const int kcol = kb + nt * 16 + l16;
        const float s  = cs[nt][r] * 0.125f;           // 1/sqrt(64)
        cs[nt][r] = (kcol <= qrow) ? s : -1e30f;
      }
    }

    // online softmax; each row lives in one 16-lane half-group
    #pragma unroll
    for (int r = 0; r < 8; r++) {
      float rmax = fmaxf(cs[0][r], cs[1][r]);
      #pragma unroll
      for (int ox = 8; ox >= 1; ox >>= 1) rmax = fmaxf(rmax, __shfl_xor(rmax, ox, 32));
      const float mnew  = fmaxf(m[r], rmax);
      const float scale = __expf(m[r] - mnew);
      const float p0 = __expf(cs[0][r] - mnew);
      const float p1 = __expf(cs[1][r] - mnew);
      cs[0][r] = p0; cs[1][r] = p1;
      float rsum = p0 + p1;
      #pragma unroll
      for (int ox = 8; ox >= 1; ox >>= 1) rsum += __shfl_xor(rsum, ox, 32);
      l[r] = l[r] * scale + rsum;
      m[r] = mnew;
      #pragma unroll
      for (int ot = 0; ot < 4; ot++) o[ot][r] = o[ot][r] * scale;
    }

    // repack P (C layout) -> A layout via per-wave LDS
    #pragma unroll
    for (int nt = 0; nt < 2; nt++)
      #pragma unroll
      for (int r = 0; r < 8; r++)
        Ps[wid][r + 8 * hf][nt * 16 + l16] = f2bf(cs[nt][r]);
    asm volatile("s_wait_dscnt 0" ::: "memory");       // wave-local LDS RAW

    const u16* pp = &Ps[wid][l16][0];
    v16bf ap = load_frag(pp + hf * 8, pp + 16 + hf * 8);
    #pragma unroll
    for (int ot = 0; ot < 4; ot++) {
      u16 tv[16];                                      // B: N=l16(d), K=key in table order
      #pragma unroll
      for (int s = 0; s < 16; s++) {
        const int kk = (s < 8) ? hf * 8 + s : 16 + hf * 8 + (s - 8);
        tv[s] = Vs[kk][ot * 16 + l16];
      }
      v16bf bvf = __builtin_bit_cast(v16bf, tv);
      o[ot] = __builtin_amdgcn_wmma_f32_16x16x32_bf16(false, ap, false, bvf, (short)0, o[ot], false, false);
    }
    __syncthreads();
  }

  // finalize: O /= l, write bf16 in (b, s, h*64+d) layout for the out-proj GEMM
  u16* Obase = Ao + (size_t)b * SEQ * DIMS + h * HD;
  #pragma unroll
  for (int ot = 0; ot < 4; ot++) {
    #pragma unroll
    for (int r = 0; r < 8; r++) {
      const int qrow = qrow0 + r + 8 * hf;
      Obase[(size_t)qrow * DIMS + ot * 16 + l16] = f2bf(o[ot][r] / l[r]);
    }
  }
}

// ---------------------------------------------------------------- launch
extern "C" void kernel_launch(void* const* d_in, const int* in_sizes, int n_in,
                              void* d_out, int out_size, void* d_ws, size_t ws_size,
                              hipStream_t stream) {
  (void)in_sizes; (void)n_in; (void)out_size; (void)ws_size;
  const float* q   = (const float*)d_in[0];
  const float* kv  = (const float*)d_in[1];
  // d_in[2] = mask (causal tril) — applied analytically in gqa_flash
  const float* Wq  = (const float*)d_in[3];
  const float* bq  = (const float*)d_in[4];
  const float* Wk  = (const float*)d_in[5];
  const float* bk  = (const float*)d_in[6];
  const float* Wv  = (const float*)d_in[7];
  const float* bv  = (const float*)d_in[8];
  const float* Wo  = (const float*)d_in[9];
  const float* bo  = (const float*)d_in[10];
  float* out = (float*)d_out;

  char* ws = (char*)d_ws;
  size_t off = 0;
  auto carve = [&](size_t elems) -> u16* {
    u16* p = (u16*)(ws + off);
    off = (off + elems * sizeof(u16) + 255) & ~(size_t)255;
    return p;
  };
  u16* qbf  = carve((size_t)BS * DIMS);
  u16* kvbf = carve((size_t)BS * DIMS);
  u16* Wqbf = carve((size_t)DIMS * DIMS);
  u16* Wkbf = carve((size_t)DIMS * KVD);
  u16* Wvbf = carve((size_t)DIMS * KVD);
  u16* Wobf = carve((size_t)DIMS * DIMS);
  u16* Qp   = carve((size_t)BS * DIMS);
  u16* Kpj  = carve((size_t)BS * KVD);
  u16* Vpj  = carve((size_t)BS * KVD);
  u16* Ao   = carve((size_t)BS * DIMS);

  const int CB = 2048, CT = 256;
  cvt_f32_bf16<<<CB, CT, 0, stream>>>(q,  qbf,  BS * DIMS);
  cvt_f32_bf16<<<CB, CT, 0, stream>>>(kv, kvbf, BS * DIMS);
  cvt_f32_bf16<<<CB, CT, 0, stream>>>(Wq, Wqbf, DIMS * DIMS);
  cvt_f32_bf16<<<CB, CT, 0, stream>>>(Wk, Wkbf, DIMS * KVD);
  cvt_f32_bf16<<<CB, CT, 0, stream>>>(Wv, Wvbf, DIMS * KVD);
  cvt_f32_bf16<<<CB, CT, 0, stream>>>(Wo, Wobf, DIMS * DIMS);

  // Q/K/V projections (bf16 out)
  gemm_bf16<<<dim3(DIMS / BN, BS / BM), 256, 0, stream>>>(
      qbf,  Wqbf, bq, nullptr, Qp,  DIMS, DIMS, DIMS, DIMS);
  gemm_bf16<<<dim3(KVD / BN, BS / BM), 256, 0, stream>>>(
      kvbf, Wkbf, bk, nullptr, Kpj, DIMS, DIMS, KVD, KVD);
  gemm_bf16<<<dim3(KVD / BN, BS / BM), 256, 0, stream>>>(
      kvbf, Wvbf, bv, nullptr, Vpj, DIMS, DIMS, KVD, KVD);

  // flash attention
  gqa_flash<<<dim3(SEQ / 128, QH, BSZ), 256, 0, stream>>>(Qp, Kpj, Vpj, Ao);

  // output projection (fp32 out -> d_out)
  gemm_bf16<<<dim3(DIMS / BN, BS / BM), 256, 0, stream>>>(
      Ao, Wobf, bo, out, nullptr, DIMS, DIMS, DIMS, DIMS);
}
